// MixtureOfExperts_83270825935008
// MI455X (gfx1250) — matrix operand order repaired and checked by reading, not compiled, not run on hardware
//
#include <hip/hip_runtime.h>

#define B_DIM   8192
#define IN_DIM  1024
#define HID_DIM 4096
#define OUT_DIM 1024
#define E_DIM   8

typedef __attribute__((ext_vector_type(16))) __bf16 v16bf;
typedef __attribute__((ext_vector_type(8)))  __bf16 v8bf;
typedef __attribute__((ext_vector_type(4)))  __bf16 v4bf;
typedef __attribute__((ext_vector_type(8)))  float  v8f;
typedef __attribute__((ext_vector_type(4)))  int    v4i;

#if __has_builtin(__builtin_amdgcn_global_load_async_to_lds_b128)
#define MOE_ASYNC 1
#else
#define MOE_ASYNC 0
#endif

// Copy 16B global -> LDS. Async (ASYNCcnt-tracked, no VGPR data) when available.
__device__ __forceinline__ void moe_copy16(const __bf16* g, __bf16* l) {
#if MOE_ASYNC
  __builtin_amdgcn_global_load_async_to_lds_b128(
      (__attribute__((address_space(1))) v4i*)(v4i*)const_cast<__bf16*>(g),
      (__attribute__((address_space(3))) v4i*)(v4i*)l, 0, 0);
#else
  *(v8bf*)l = *(const v8bf*)g;
#endif
}

__device__ __forceinline__ void moe_wait_async() {
#if MOE_ASYNC
#if __has_builtin(__builtin_amdgcn_s_wait_asynccnt)
  __builtin_amdgcn_s_wait_asynccnt(0);
#else
  asm volatile("s_wait_asynccnt 0x0" ::: "memory");
#endif
#endif
}

// ---------------- f32 -> bf16 elementwise convert (vectorized x4) ----------
__global__ void moe_cvt_f32_bf16(const float* __restrict__ s,
                                 __bf16* __restrict__ d, long n4) {
  long i = (long)blockIdx.x * blockDim.x + threadIdx.x;
  long stride = (long)gridDim.x * blockDim.x;
  for (; i < n4; i += stride) {
    float4 f = ((const float4*)s)[i];
    v4bf o;
    o[0] = (__bf16)f.x; o[1] = (__bf16)f.y; o[2] = (__bf16)f.z; o[3] = (__bf16)f.w;
    ((v4bf*)d)[i] = o;
  }
}

// ------------- f32 [R][C] -> bf16 [C][R] transpose+convert (LDS tiled) -----
__global__ void moe_transpose_cvt(const float* __restrict__ src,
                                  __bf16* __restrict__ dst, int R, int C) {
  __shared__ float tile[32][33];
  size_t base = (size_t)blockIdx.z * R * C;
  src += base; dst += base;
  int c0 = blockIdx.x * 32, r0 = blockIdx.y * 32;
  int tx = threadIdx.x & 31, ty = threadIdx.x >> 5;  // 32x8 threads
#pragma unroll
  for (int j = 0; j < 32; j += 8)
    tile[ty + j][tx] = src[(size_t)(r0 + ty + j) * C + c0 + tx];
  __syncthreads();
#pragma unroll
  for (int j = 0; j < 32; j += 8)
    dst[(size_t)(c0 + ty + j) * R + r0 + tx] = (__bf16)tile[tx][ty + j];
}

// ---------------- gate = softmax(x @ Wg + bg) over E=8 ---------------------
__global__ void moe_gate(const float* __restrict__ x, const float* __restrict__ Wg,
                         const float* __restrict__ bg, float* __restrict__ gate) {
  int lane = threadIdx.x & 31;
  int wave = threadIdx.x >> 5;
  int b = blockIdx.x * 8 + wave;          // one wave32 per row
  float acc[E_DIM];
#pragma unroll
  for (int e = 0; e < E_DIM; ++e) acc[e] = 0.f;
  const float* xr = x + (size_t)b * IN_DIM;
  for (int i = lane; i < IN_DIM; i += 32) {
    float xv = xr[i];
    const float* wr = Wg + (size_t)i * E_DIM;
#pragma unroll
    for (int e = 0; e < E_DIM; ++e) acc[e] += xv * wr[e];
  }
#pragma unroll
  for (int e = 0; e < E_DIM; ++e) {
#pragma unroll
    for (int m = 16; m >= 1; m >>= 1) acc[e] += __shfl_xor(acc[e], m, 32);
    acc[e] += bg[e];
  }
  if (lane == 0) {
    float mx = acc[0];
#pragma unroll
    for (int e = 1; e < E_DIM; ++e) mx = fmaxf(mx, acc[e]);
    float s = 0.f;
#pragma unroll
    for (int e = 0; e < E_DIM; ++e) { acc[e] = __expf(acc[e] - mx); s += acc[e]; }
    float inv = 1.f / s;
#pragma unroll
    for (int e = 0; e < E_DIM; ++e) gate[(size_t)b * E_DIM + e] = acc[e] * inv;
  }
}

// ---------------- bf16 WMMA GEMM: C = A[M,K] * Bt[N,K]^T + bias ------------
// Block tile 256x128, BK=32, 8 waves (4Mx2N), wave tile 64x64 (16 WMMA/kstep),
// double-buffered async global->LDS staging.
// MODE 0: H = relu(C) stored bf16 (outB)
// MODE 1: outF[m, n] (+)= gate[m,e] * C   (f32)
template <int MODE>
__global__ __launch_bounds__(256)
void moe_gemm_bf16(const __bf16* __restrict__ A, const __bf16* __restrict__ Bt,
                   const float* __restrict__ bias,
                   __bf16* __restrict__ outB, float* __restrict__ outF,
                   const float* __restrict__ gate, int Ktot, int Ntot,
                   int expert, int accum) {
  __shared__ __bf16 As[2][256 * 32];   // 16 KB per buffer
  __shared__ __bf16 Bs[2][128 * 32];   //  8 KB per buffer
  const int tid   = threadIdx.x;
  const int lane  = tid & 31;
  const int wave  = tid >> 5;
  const int waveM = wave >> 1;             // 0..3 -> 64 rows each
  const int waveN = wave & 1;              // 0..1 -> 64 cols each
  const int lm    = lane & 15;
  const int kA0   = (lane < 16) ? 0 : 8;   // A frag: chunks {kA0, kA0+16}
  const int kB0   = (lane < 16) ? 0 : 8;   // B frag: chunks {kB0, kB0+16}
  const int m0 = blockIdx.y * 256;
  const int n0 = blockIdx.x * 128;

  // per-thread staging chunks (8 bf16 = 16B each): A 1024 chunks, B 512 chunks
  int offA[4]; const __bf16* gA[4];
#pragma unroll
  for (int j = 0; j < 4; ++j) {
    int c = tid + j * 256;
    int row = c >> 2, kc = (c & 3) * 8;
    offA[j] = row * 32 + kc;
    gA[j] = A + (size_t)(m0 + row) * Ktot + kc;
  }
  int offB[2]; const __bf16* gB[2];
#pragma unroll
  for (int j = 0; j < 2; ++j) {
    int c = tid + j * 256;
    int row = c >> 2, kc = (c & 3) * 8;
    offB[j] = row * 32 + kc;
    gB[j] = Bt + (size_t)(n0 + row) * Ktot + kc;
  }

  v8f acc[4][4];
#pragma unroll
  for (int mi = 0; mi < 4; ++mi)
#pragma unroll
    for (int ni = 0; ni < 4; ++ni)
      acc[mi][ni] = (v8f){0.f, 0.f, 0.f, 0.f, 0.f, 0.f, 0.f, 0.f};

  const int nk = Ktot >> 5;
  // prologue: stage k-tile 0 into buffer 0
#pragma unroll
  for (int j = 0; j < 4; ++j) moe_copy16(gA[j], &As[0][offA[j]]);
#pragma unroll
  for (int j = 0; j < 2; ++j) moe_copy16(gB[j], &Bs[0][offB[j]]);

  for (int ki = 0; ki < nk; ++ki) {
    const int cur = ki & 1;
    moe_wait_async();        // staged copies for buf[cur] complete (this wave)
    __syncthreads();         // all waves done with buf[cur^1] reads; buf[cur] ready

    if (ki + 1 < nk) {       // stage next tile into other buffer (overlaps wmma)
      const int kn = (ki + 1) << 5;
#pragma unroll
      for (int j = 0; j < 4; ++j) moe_copy16(gA[j] + kn, &As[cur ^ 1][offA[j]]);
#pragma unroll
      for (int j = 0; j < 2; ++j) moe_copy16(gB[j] + kn, &Bs[cur ^ 1][offB[j]]);
    }
    if (ki + 2 < nk) {       // nudge k+2 tile toward L2
      const int kp = (ki + 2) << 5;
      __builtin_prefetch(gA[0] + kp, 0, 1);
      __builtin_prefetch(gB[0] + kp, 0, 1);
    }

    v16bf aF[4], bF[4];
#pragma unroll
    for (int mi = 0; mi < 4; ++mi) {
      const __bf16* p = &As[cur][(waveM * 64 + mi * 16 + lm) * 32];
      union { v16bf v; v8bf h[2]; } u;
      u.h[0] = *(const v8bf*)(p + kA0);
      u.h[1] = *(const v8bf*)(p + kA0 + 16);
      aF[mi] = u.v;
    }
#pragma unroll
    for (int ni = 0; ni < 4; ++ni) {
      const __bf16* p = &Bs[cur][(waveN * 64 + ni * 16 + lm) * 32];
      union { v16bf v; v8bf h[2]; } u;
      u.h[0] = *(const v8bf*)(p + kB0);
      u.h[1] = *(const v8bf*)(p + kB0 + 16);
      bF[ni] = u.v;
    }
#pragma unroll
    for (int mi = 0; mi < 4; ++mi)
#pragma unroll
      for (int ni = 0; ni < 4; ++ni)
        acc[mi][ni] = __builtin_amdgcn_wmma_f32_16x16x32_bf16(
            false, aF[mi], false, bF[ni], (short)0, acc[mi][ni], false, false);
  }

  // epilogue: lane holds N = lm, M = (lane<16 ? 0 : 8) + vgpr
  const int mBase = m0 + waveM * 64 + ((lane < 16) ? 0 : 8);
#pragma unroll
  for (int ni = 0; ni < 4; ++ni) {
    int n = n0 + waveN * 64 + ni * 16 + lm;
    float bv = bias[n];
#pragma unroll
    for (int mi = 0; mi < 4; ++mi) {
#pragma unroll
      for (int v = 0; v < 8; ++v) {
        int m = mBase + mi * 16 + v;
        float val = acc[mi][ni][v] + bv;
        if (MODE == 0) {
          outB[(size_t)m * Ntot + n] = (__bf16)fmaxf(val, 0.f);
        } else {
          float g = gate[(size_t)m * E_DIM + expert];
          float* o = outF + (size_t)m * Ntot + n;
          float prev = accum ? *o : 0.f;
          *o = prev + g * val;
        }
      }
    }
  }
}

extern "C" void kernel_launch(void* const* d_in, const int* in_sizes, int n_in,
                              void* d_out, int out_size, void* d_ws, size_t ws_size,
                              hipStream_t stream) {
  (void)in_sizes; (void)n_in; (void)out_size; (void)ws_size;
  const float* x  = (const float*)d_in[0];
  const float* W1 = (const float*)d_in[1];
  const float* b1 = (const float*)d_in[2];
  const float* W2 = (const float*)d_in[3];
  const float* b2 = (const float*)d_in[4];
  const float* Wg = (const float*)d_in[5];
  const float* bg = (const float*)d_in[6];
  float* out = (float*)d_out;

  char* ws = (char*)d_ws;
  __bf16* xb   = (__bf16*)(ws);                                  // 16 MiB
  __bf16* W1T  = (__bf16*)(ws + (size_t)16  * 1024 * 1024);      // 64 MiB [E][HID][IN]
  __bf16* W2T  = (__bf16*)(ws + (size_t)80  * 1024 * 1024);      // 64 MiB [E][OUT][HID]
  __bf16* H    = (__bf16*)(ws + (size_t)144 * 1024 * 1024);      // 64 MiB [B][HID]
  float*  gate = (float*) (ws + (size_t)208 * 1024 * 1024);      // 256 KiB

  moe_cvt_f32_bf16<<<2048, 256, 0, stream>>>(x, xb, (long)B_DIM * IN_DIM / 4);
  moe_transpose_cvt<<<dim3(HID_DIM / 32, IN_DIM / 32, E_DIM), 256, 0, stream>>>(
      W1, W1T, IN_DIM, HID_DIM);
  moe_transpose_cvt<<<dim3(OUT_DIM / 32, HID_DIM / 32, E_DIM), 256, 0, stream>>>(
      W2, W2T, HID_DIM, OUT_DIM);
  moe_gate<<<B_DIM / 8, 256, 0, stream>>>(x, Wg, bg, gate);

  for (int e = 0; e < E_DIM; ++e) {
    moe_gemm_bf16<0><<<dim3(HID_DIM / 128, B_DIM / 256), 256, 0, stream>>>(
        xb, W1T + (size_t)e * IN_DIM * HID_DIM, b1 + (size_t)e * HID_DIM,
        H, nullptr, nullptr, IN_DIM, HID_DIM, e, 0);
    moe_gemm_bf16<1><<<dim3(OUT_DIM / 128, B_DIM / 256), 256, 0, stream>>>(
        H, W2T + (size_t)e * HID_DIM * OUT_DIM, b2 + (size_t)e * OUT_DIM,
        nullptr, out, gate, HID_DIM, OUT_DIM, e, e > 0);
  }
}
